// MultiheadAttention_72567767433949
// MI455X (gfx1250) — matrix operand order repaired and checked by reading, not compiled, not run on hardware
//
#include <hip/hip_runtime.h>
#include <hip/hip_bf16.h>
#include <math.h>

#define B_  2
#define L_  2048
#define DM_ 1024   // DMODEL == H_*DV_
#define H_  16
#define DK_ 64
#define DV_ 64

typedef __attribute__((ext_vector_type(16))) __bf16 v16bf;
typedef __attribute__((ext_vector_type(8)))  float  v8f;
typedef __attribute__((ext_vector_type(4)))  unsigned int u32x4;
typedef __attribute__((ext_vector_type(8)))  int i32x8;
typedef __attribute__((ext_vector_type(4)))  int i32x4;

#if __has_builtin(__builtin_amdgcn_tensor_load_to_lds)
#define HAVE_TDM 1
#else
#define HAVE_TDM 0
#endif

// ---------------------------------------------------------------------------
// WMMA helpers (CDNA5 wave32, 16x16x32 bf16 -> f32)
// ---------------------------------------------------------------------------
__device__ __forceinline__ v8f wmma_bf16(v16bf a, v16bf b, v8f c) {
  return __builtin_amdgcn_wmma_f32_16x16x32_bf16(false, a, false, b,
                                                 (short)0, c, false, false);
}

// A fragment: 16x32 (MxK) bf16, row-major source, 'stride' elems/row.
// lanes 0-15: row M=lane, K={0..7,16..23}; lanes 16-31: row M=lane-16, K={8..15,24..31}
__device__ __forceinline__ v16bf load_a(const __bf16* base, int stride, int lane) {
  const __bf16* p = base + (size_t)(lane & 15) * stride + ((lane & 16) ? 8 : 0);
  union { v16bf v; uint4 u[2]; } f;
  f.u[0] = *(const uint4*)(p);
  f.u[1] = *(const uint4*)(p + 16);
  return f.v;
}

// B fragment: 32x16 (KxN) bf16 from "B^T" storage (row n = column n of B).
// lanes 0-15: col N=lane, K={0..7,16..23}; lanes 16-31: col N=lane-16, K={8..15,24..31}
__device__ __forceinline__ v16bf load_b(const __bf16* baseT, int stride, int lane) {
  const __bf16* p = baseT + (size_t)(lane & 15) * stride + ((lane & 16) ? 8 : 0);
  union { v16bf v; uint4 u[2]; } f;
  f.u[0] = *(const uint4*)(p);
  f.u[1] = *(const uint4*)(p + 16);
  return f.v;
}

// ---------------------------------------------------------------------------
// Tensor Data Mover: DMA a 2D bf16 tile (tile1 rows x tile0 cols) into LDS.
// D# packing per CDNA5 ISA.
// ---------------------------------------------------------------------------
__device__ __forceinline__ unsigned lds_off_of(const void* p) {
  // LDS aperture keeps the byte offset in addr[31:0] (ISA §10.2)
  return (unsigned)(unsigned long long)p;
}

#if HAVE_TDM
__device__ __forceinline__ void tdm_load_2d(const void* gsrc, unsigned lds,
                                            unsigned dim0, unsigned dim1,
                                            unsigned stride0,
                                            unsigned tile0, unsigned tile1) {
  unsigned long long ga = (unsigned long long)gsrc;
  u32x4 g0;
  g0[0] = 1u;                                        // count=1, user mode
  g0[1] = lds;                                       // lds_addr
  g0[2] = (unsigned)(ga & 0xffffffffull);            // global_addr[31:0]
  g0[3] = (unsigned)((ga >> 32) & 0x01ffffffull) | (2u << 30);  // addr[56:32] | type=2
  i32x8 g1;
  g1[0] = (int)(1u << 16);                           // data_size=1 (2 bytes)
  g1[1] = (int)((dim0 & 0xffffu) << 16);             // tensor_dim0[15:0]
  g1[2] = (int)((dim0 >> 16) | ((dim1 & 0xffffu) << 16));      // dim0 hi | dim1 lo
  g1[3] = (int)((dim1 >> 16) | (tile0 << 16));       // dim1 hi | tile_dim0
  g1[4] = (int)(tile1 & 0xffffu);                    // tile_dim1 | tile_dim2=0
  g1[5] = (int)stride0;                              // tensor_dim0_stride[31:0]
  g1[6] = 0;
  g1[7] = 0;
  i32x4 z4 = {0, 0, 0, 0};
#if defined(__clang_major__) && __clang_major__ >= 23
  i32x8 z8 = {0, 0, 0, 0, 0, 0, 0, 0};
  __builtin_amdgcn_tensor_load_to_lds(g0, g1, z4, z4, z8, 0);
#else
  __builtin_amdgcn_tensor_load_to_lds(g0, g1, z4, z4, 0);
#endif
}
#else
__device__ __forceinline__ void stage_tile_fallback(const __bf16* g, __bf16* l,
                                                    int rows, int cols, int gstride,
                                                    int lane) {
  for (int i = lane * 8; i < rows * cols; i += 32 * 8) {
    int r = i / cols, c = i - r * cols;
    *(uint4*)(l + i) = *(const uint4*)(g + (size_t)r * gstride + c);
  }
}
#endif

// ---------------------------------------------------------------------------
// Stage 0: dtype conversion / weight transposition
// ---------------------------------------------------------------------------
__global__ void cvt_f32_to_bf16(const float* __restrict__ s, __bf16* __restrict__ d, int n) {
  int i = blockIdx.x * blockDim.x + threadIdx.x;
  if (i < n) d[i] = (__bf16)s[i];
}

// proj [H, DMODEL, DK] (f32) -> projT [H, DK, DMODEL] (bf16)
__global__ void transpose_proj(const float* __restrict__ p, __bf16* __restrict__ t, int n) {
  int i = blockIdx.x * blockDim.x + threadIdx.x;
  if (i >= n) return;
  int h   = i / (DM_ * DK_);
  int rem = i - h * (DM_ * DK_);
  int d   = rem / DK_;
  int e   = rem - d * DK_;
  t[((size_t)h * DK_ + e) * DM_ + d] = (__bf16)p[i];
}

// ---------------------------------------------------------------------------
// Stage 1: per-head projections (ping-pong pipelined K loop, no reg rotation)
// ---------------------------------------------------------------------------
__global__ __launch_bounds__(128)
void proj_kernel(const __bf16* __restrict__ qbf, const __bf16* __restrict__ kbf,
                 const __bf16* __restrict__ vbf,
                 const __bf16* __restrict__ wqT, const __bf16* __restrict__ wkT,
                 const __bf16* __restrict__ wvT,
                 const float* __restrict__ bq, const float* __restrict__ bk,
                 const float* __restrict__ bv,
                 __bf16* __restrict__ qh, __bf16* __restrict__ kh,
                 __bf16* __restrict__ vhT) {
  const int lane  = threadIdx.x & 31;
  const int wid   = threadIdx.x >> 5;
  const int which = blockIdx.z;
  const int bh    = blockIdx.y;
  const int b     = bh / H_;
  const int h     = bh % H_;
  const int mt    = blockIdx.x * 4 + wid;

  const __bf16* X    = (which == 0) ? qbf : (which == 1) ? kbf : vbf;
  const __bf16* WT   = (which == 0) ? wqT : (which == 1) ? wkT : wvT;
  const float*  bias = (which == 0) ? bq  : (which == 1) ? bk  : bv;

  const __bf16* Xbase = X  + ((size_t)b * L_ + (size_t)mt * 16) * DM_;
  const __bf16* Wbase = WT + (size_t)h * DK_ * DM_;

  v8f acc[4] = {v8f{}, v8f{}, v8f{}, v8f{}};

  // ping-pong register sets: set0 holds k0, set1 holds k0+32
  v16bf a0 = load_a(Xbase, DM_, lane);
  v16bf b0[4], b1[4];
  v16bf a1;
#pragma unroll
  for (int n = 0; n < 4; ++n) b0[n] = load_b(Wbase + (size_t)(n * 16) * DM_, DM_, lane);

  int k0 = 0;
  for (; k0 < DM_ - 64; k0 += 64) {
    a1 = load_a(Xbase + k0 + 32, DM_, lane);
#pragma unroll
    for (int n = 0; n < 4; ++n)
      b1[n] = load_b(Wbase + (size_t)(n * 16) * DM_ + k0 + 32, DM_, lane);
#pragma unroll
    for (int n = 0; n < 4; ++n) acc[n] = wmma_bf16(a0, b0[n], acc[n]);

    a0 = load_a(Xbase + k0 + 64, DM_, lane);
#pragma unroll
    for (int n = 0; n < 4; ++n)
      b0[n] = load_b(Wbase + (size_t)(n * 16) * DM_ + k0 + 64, DM_, lane);
#pragma unroll
    for (int n = 0; n < 4; ++n) acc[n] = wmma_bf16(a1, b1[n], acc[n]);
  }
  // tail: k0 == DM_-64 -> steps k0 and k0+32
  a1 = load_a(Xbase + k0 + 32, DM_, lane);
#pragma unroll
  for (int n = 0; n < 4; ++n)
    b1[n] = load_b(Wbase + (size_t)(n * 16) * DM_ + k0 + 32, DM_, lane);
#pragma unroll
  for (int n = 0; n < 4; ++n) acc[n] = wmma_bf16(a0, b0[n], acc[n]);
#pragma unroll
  for (int n = 0; n < 4; ++n) acc[n] = wmma_bf16(a1, b1[n], acc[n]);

  const int col   = lane & 15;
  const int rbase = (lane & 16) ? 8 : 0;
  const float scale = (which == 0) ? 0.125f : 1.0f;  // fold 1/sqrt(64) into qh

  if (which != 2) {
    __bf16* O = ((which == 0) ? qh : kh) + ((size_t)bh * L_ + (size_t)mt * 16) * DK_;
#pragma unroll
    for (int n = 0; n < 4; ++n) {
      float bb = bias[h * DK_ + n * 16 + col];
#pragma unroll
      for (int r = 0; r < 8; ++r) {
        float v = (acc[n][r] + bb) * scale;
        O[(size_t)(rbase + r) * DK_ + n * 16 + col] = (__bf16)v;
      }
    }
  } else {
    __bf16* O = vhT + (size_t)bh * DV_ * L_;
#pragma unroll
    for (int n = 0; n < 4; ++n) {
      float bb = bias[h * DV_ + n * 16 + col];
#pragma unroll
      for (int r = 0; r < 8; ++r) {
        float v = acc[n][r] + bb;
        O[(size_t)(n * 16 + col) * L_ + (size_t)mt * 16 + rbase + r] = (__bf16)v;
      }
    }
  }
}

// ---------------------------------------------------------------------------
// Stage 2: flash attention with TDM-staged, double-buffered K/V tiles in LDS.
// One wave = 16 queries of one (b,h); 32-key blocks, online softmax.
// ---------------------------------------------------------------------------
__global__ __launch_bounds__(128)
void attn_kernel(const __bf16* __restrict__ qh, const __bf16* __restrict__ kh,
                 const __bf16* __restrict__ vhT, __bf16* __restrict__ hid) {
  __shared__ alignas(16) __bf16 Kl[4][2][32 * 64];  // K tile: 32 keys x 64 dk
  __shared__ alignas(16) __bf16 Vl[4][2][64 * 32];  // V^T tile: 64 dv x 32 keys
  __shared__ alignas(16) __bf16 Pl[4][16 * 32];     // P relayout buffer

  const int lane = threadIdx.x & 31;
  const int wid  = threadIdx.x >> 5;
  const int bh   = blockIdx.y;
  const int b    = bh / H_;
  const int h    = bh % H_;
  const int mt   = blockIdx.x * 4 + wid;

  const __bf16* Qb = qh  + ((size_t)bh * L_ + (size_t)mt * 16) * DK_;
  const __bf16* Kb = kh  + (size_t)bh * L_ * DK_;
  const __bf16* Vb = vhT + (size_t)bh * DV_ * L_;
  __bf16* P = Pl[wid];

  // kick off DMA of block 0 before anything else
#if HAVE_TDM
  tdm_load_2d(Kb, lds_off_of(&Kl[wid][0][0]), DK_, L_, DK_, 64, 32);
  tdm_load_2d(Vb, lds_off_of(&Vl[wid][0][0]), L_, DV_, L_, 32, 64);
#else
  stage_tile_fallback(Kb, &Kl[wid][0][0], 32, 64, DK_, lane);
  stage_tile_fallback(Vb, &Vl[wid][0][0], 64, 32, L_, lane);
#endif

  const v16bf aq0 = load_a(Qb,      DK_, lane);   // dk 0..31
  const v16bf aq1 = load_a(Qb + 32, DK_, lane);   // dk 32..63

  const int col   = lane & 15;
  const int rbase = (lane & 16) ? 8 : 0;

  v8f accO[4] = {v8f{}, v8f{}, v8f{}, v8f{}};
  float mrow[8], lrow[8];
#pragma unroll
  for (int r = 0; r < 8; ++r) { mrow[r] = -1e30f; lrow[r] = 0.0f; }

  int cur = 0;
  for (int j = 0; j < L_; j += 32) {
    // wait for the current buffers' DMA, then immediately start the next one
#if HAVE_TDM
    __builtin_amdgcn_s_wait_tensorcnt(0);
#else
    asm volatile("s_wait_dscnt 0" ::: "memory");
#endif
    asm volatile("" ::: "memory");
    __builtin_amdgcn_wave_barrier();

    if (j + 32 < L_) {
      const int nxt = cur ^ 1;
#if HAVE_TDM
      tdm_load_2d(Kb + (size_t)(j + 32) * DK_, lds_off_of(&Kl[wid][nxt][0]),
                  DK_, L_, DK_, 64, 32);
      tdm_load_2d(Vb + (j + 32), lds_off_of(&Vl[wid][nxt][0]),
                  L_, DV_, L_, 32, 64);
#else
      stage_tile_fallback(Kb + (size_t)(j + 32) * DK_, &Kl[wid][nxt][0], 32, 64, DK_, lane);
      stage_tile_fallback(Vb + (j + 32), &Vl[wid][nxt][0], 64, 32, L_, lane);
#endif
    }

    const __bf16* KL = Kl[wid][cur];
    const __bf16* VL = Vl[wid][cur];

    // S tiles: 16 queries x 32 keys, contraction dk=64 (B-operand from LDS)
    v8f s0 = v8f{}, s1 = v8f{};
    s0 = wmma_bf16(aq0, load_b(KL,                64, lane), s0);
    s0 = wmma_bf16(aq1, load_b(KL + 32,           64, lane), s0);
    s1 = wmma_bf16(aq0, load_b(KL + 16 * 64,      64, lane), s1);
    s1 = wmma_bf16(aq1, load_b(KL + 16 * 64 + 32, 64, lane), s1);

    // row max across this block (cross-lane within 16-lane halves; wave32)
    float rmax[8];
#pragma unroll
    for (int r = 0; r < 8; ++r) rmax[r] = fmaxf(s0[r], s1[r]);
#pragma unroll
    for (int m = 8; m >= 1; m >>= 1)
#pragma unroll
      for (int r = 0; r < 8; ++r) rmax[r] = fmaxf(rmax[r], __shfl_xor(rmax[r], m));

    float alpha[8];
#pragma unroll
    for (int r = 0; r < 8; ++r) {
      float mn = fmaxf(mrow[r], rmax[r]);
      alpha[r] = __expf(mrow[r] - mn);
      mrow[r]  = mn;
    }

    float rsum[8];
#pragma unroll
    for (int r = 0; r < 8; ++r) {
      float p0 = __expf(s0[r] - mrow[r]);
      float p1 = __expf(s1[r] - mrow[r]);
      rsum[r] = p0 + p1;
      P[(rbase + r) * 32 + col]      = (__bf16)p0;
      P[(rbase + r) * 32 + col + 16] = (__bf16)p1;
    }
#pragma unroll
    for (int m = 8; m >= 1; m >>= 1)
#pragma unroll
      for (int r = 0; r < 8; ++r) rsum[r] += __shfl_xor(rsum[r], m);
#pragma unroll
    for (int r = 0; r < 8; ++r) lrow[r] = lrow[r] * alpha[r] + rsum[r];
#pragma unroll
    for (int t = 0; t < 4; ++t)
#pragma unroll
      for (int r = 0; r < 8; ++r) accO[t][r] *= alpha[r];

    // cross-lane LDS dependency inside the wave for the P relayout
    asm volatile("s_wait_dscnt 0" ::: "memory");
    __builtin_amdgcn_wave_barrier();

    v16bf pa = load_a(P, 32, lane);
#pragma unroll
    for (int t = 0; t < 4; ++t) {
      v16bf bv = load_b(VL + (size_t)(t * 16) * 32, 32, lane);
      accO[t] = wmma_bf16(pa, bv, accO[t]);
    }
    cur ^= 1;
  }

  // epilogue: normalize, write head-major hid[b, l, h*DV + dv] (bf16)
  __bf16* O = hid + ((size_t)b * L_ + (size_t)mt * 16) * DM_ + (size_t)h * DV_;
  float inv[8];
#pragma unroll
  for (int r = 0; r < 8; ++r) inv[r] = 1.0f / lrow[r];
#pragma unroll
  for (int t = 0; t < 4; ++t)
#pragma unroll
    for (int r = 0; r < 8; ++r)
      O[(size_t)(rbase + r) * DM_ + t * 16 + col] = (__bf16)(accO[t][r] * inv[r]);
}

// ---------------------------------------------------------------------------
// Stage 3: out = hid @ w_out^T  (ping-pong pipelined)
// ---------------------------------------------------------------------------
__global__ __launch_bounds__(128)
void out_kernel(const __bf16* __restrict__ hid, const __bf16* __restrict__ wout,
                float* __restrict__ out) {
  const int lane = threadIdx.x & 31;
  const int wid  = threadIdx.x >> 5;
  const int mt   = blockIdx.x;
  const int nt   = blockIdx.y * 4 + wid;

  const __bf16* A  = hid  + (size_t)mt * 16 * DM_;
  const __bf16* Bt = wout + (size_t)nt * 16 * DM_;

  v8f acc = v8f{};
  v16bf a0 = load_a(A, DM_, lane);
  v16bf b0 = load_b(Bt, DM_, lane);
  v16bf a1, b1;
  int k0 = 0;
  for (; k0 < DM_ - 64; k0 += 64) {
    a1 = load_a(A + k0 + 32, DM_, lane);
    b1 = load_b(Bt + k0 + 32, DM_, lane);
    acc = wmma_bf16(a0, b0, acc);
    a0 = load_a(A + k0 + 64, DM_, lane);
    b0 = load_b(Bt + k0 + 64, DM_, lane);
    acc = wmma_bf16(a1, b1, acc);
  }
  a1 = load_a(A + k0 + 32, DM_, lane);
  b1 = load_b(Bt + k0 + 32, DM_, lane);
  acc = wmma_bf16(a0, b0, acc);
  acc = wmma_bf16(a1, b1, acc);

  const int col   = lane & 15;
  const int rbase = (lane & 16) ? 8 : 0;
  float* O = out + (size_t)mt * 16 * DM_ + (size_t)nt * 16;
#pragma unroll
  for (int r = 0; r < 8; ++r) O[(size_t)(rbase + r) * DM_ + col] = acc[r];
}

// ---------------------------------------------------------------------------
// Host launcher
// ---------------------------------------------------------------------------
extern "C" void kernel_launch(void* const* d_in, const int* in_sizes, int n_in,
                              void* d_out, int out_size, void* d_ws, size_t ws_size,
                              hipStream_t stream) {
  const float* q      = (const float*)d_in[0];
  const float* k      = (const float*)d_in[1];
  const float* v      = (const float*)d_in[2];
  const float* proj_q = (const float*)d_in[3];
  const float* proj_k = (const float*)d_in[4];
  const float* proj_v = (const float*)d_in[5];
  const float* bias_q = (const float*)d_in[6];
  const float* bias_k = (const float*)d_in[7];
  const float* bias_v = (const float*)d_in[8];
  const float* w_out  = (const float*)d_in[9];
  float* out = (float*)d_out;

  char* ws = (char*)d_ws;
  const size_t MB = 1u << 20;
  __bf16* qbf    = (__bf16*)(ws + 0 * MB);    // 8 MB  [B,L,DM]
  __bf16* kbf    = (__bf16*)(ws + 8 * MB);    // 8 MB
  __bf16* vbf    = (__bf16*)(ws + 16 * MB);   // 8 MB
  __bf16* wqT    = (__bf16*)(ws + 24 * MB);   // 2 MB  [H,DK,DM]
  __bf16* wkT    = (__bf16*)(ws + 26 * MB);   // 2 MB
  __bf16* wvT    = (__bf16*)(ws + 28 * MB);   // 2 MB
  __bf16* woutbf = (__bf16*)(ws + 30 * MB);   // 2 MB  [DM, H*DV]
  __bf16* qh     = (__bf16*)(ws + 32 * MB);   // 8 MB  [B,H,L,DK]
  __bf16* kh     = (__bf16*)(ws + 40 * MB);   // 8 MB
  __bf16* vhT    = (__bf16*)(ws + 48 * MB);   // 8 MB  [B,H,DV,L]
  __bf16* hid    = (__bf16*)(ws + 56 * MB);   // 8 MB  [B,L,H*DV]

  const int nact = B_ * L_ * DM_;
  const int nprj = H_ * DM_ * DK_;
  const int nwo  = DM_ * H_ * DV_;

  cvt_f32_to_bf16<<<(nact + 255) / 256, 256, 0, stream>>>(q, qbf, nact);
  cvt_f32_to_bf16<<<(nact + 255) / 256, 256, 0, stream>>>(k, kbf, nact);
  cvt_f32_to_bf16<<<(nact + 255) / 256, 256, 0, stream>>>(v, vbf, nact);
  transpose_proj<<<(nprj + 255) / 256, 256, 0, stream>>>(proj_q, wqT, nprj);
  transpose_proj<<<(nprj + 255) / 256, 256, 0, stream>>>(proj_k, wkT, nprj);
  transpose_proj<<<(nprj + 255) / 256, 256, 0, stream>>>(proj_v, wvT, nprj);
  cvt_f32_to_bf16<<<(nwo + 255) / 256, 256, 0, stream>>>(w_out, woutbf, nwo);

  proj_kernel<<<dim3(L_ / 64, B_ * H_, 3), 128, 0, stream>>>(
      qbf, kbf, vbf, wqT, wkT, wvT, bias_q, bias_k, bias_v, qh, kh, vhT);

  attn_kernel<<<dim3(L_ / 64, B_ * H_), 128, 0, stream>>>(qh, kh, vhT, hid);

  out_kernel<<<dim3(B_ * L_ / 16, DM_ / 64), 128, 0, stream>>>(hid, woutbf, out);
}
